// SlotAttentionFG_11252814316137
// MI455X (gfx1250) — compile-verified, matrix operand order checked by reading, not run on hardware
//
#include <hip/hip_runtime.h>
#include <hip/hip_bf16.h>
#include <math.h>

// Problem constants (match reference)
#define B_   16
#define N_   16384
#define C_   128
#define K_   8
#define D_   64
#define H_   128
#define ITERS_ 3
#define EPS_    1e-8f
#define LN_EPS_ 1e-5f
#define SCALE_  0.125f   // D^-0.5

// Attention-phase tiling: columns of N handled per (single-wave) workgroup
#define NPW_   256
#define CHUNKS_ (NPW_/32)

typedef __attribute__((ext_vector_type(16))) _Float16 v16h;
typedef __attribute__((ext_vector_type(8)))  _Float16 v8h;
typedef __attribute__((ext_vector_type(8)))  float    v8f;
typedef __attribute__((ext_vector_type(4)))  float    v4f;

union V16 { v16h v; v8h h[2]; };

__device__ __forceinline__ v8f zero8() {
  v8f z = {0.f,0.f,0.f,0.f,0.f,0.f,0.f,0.f};
  return z;
}

// ---------------------------------------------------------------------------
// Init: slot = mu + exp(logsigma)*noise ; convert Wk/Wv to f16
// ---------------------------------------------------------------------------
__global__ __launch_bounds__(256) void init_kernel(
    const float* __restrict__ noise, const float* __restrict__ mu,
    const float* __restrict__ logsig, const float* __restrict__ Wk,
    const float* __restrict__ Wv, float* __restrict__ slot,
    _Float16* __restrict__ Wkh, _Float16* __restrict__ Wvh)
{
  int i = blockIdx.x * blockDim.x + threadIdx.x;   // 8192 threads
  if (i < B_*K_*D_) {
    int d = i % D_;
    slot[i] = mu[d] + __expf(logsig[d]) * noise[i];
  }
  if (i < D_*C_) {
    Wkh[i] = (_Float16)Wk[i];
    Wvh[i] = (_Float16)Wv[i];
  }
}

__global__ __launch_bounds__(256) void zero_us_kernel(float* __restrict__ U,
                                                      float* __restrict__ S)
{
  int i = blockIdx.x * blockDim.x + threadIdx.x;
  if (i < B_*K_*D_) U[i] = 0.f;
  if (i < B_*K_)    S[i] = 0.f;
}

// ---------------------------------------------------------------------------
// Fused LayerNorm(feat) + k/v projection via WMMA f16->f32.
// One wave per 16 feat rows. feat is streamed with NT hints (read-once,
// 134 MB -> keep it out of L2 so the 64 MB k/v working set stays resident
// across all 3 iterations). k stored (B,N,D) f16; v stored transposed
// (B,D,N) f16 so the per-iteration B-operand loads are contiguous.
// ---------------------------------------------------------------------------
__global__ __launch_bounds__(32) void lnproj_kernel(
    const float* __restrict__ feat, const float* __restrict__ lnw,
    const float* __restrict__ lnb, const _Float16* __restrict__ Wkh,
    const _Float16* __restrict__ Wvh, _Float16* __restrict__ kbuf,
    _Float16* __restrict__ vtbuf)
{
  __shared__ __align__(16) _Float16 fe[16][C_];   // normalized feat tile (f16)
  __shared__ __align__(16) _Float16 kt[16][D_];   // k tile stage (n-major)
  __shared__ __align__(16) _Float16 vt[D_][16];   // v tile stage (d-major)
  __shared__ float wsh[C_], bsh[C_];

  const int lane = threadIdx.x;
  const int b  = blockIdx.x / (N_/16);
  const int n0 = (blockIdx.x % (N_/16)) * 16;

  for (int i = lane; i < C_; i += 32) { wsh[i] = lnw[i]; bsh[i] = lnb[i]; }

  // --- LayerNorm: 2 lanes per row, 64 cols each, kept in registers ---
  const int row  = lane >> 1;
  const int half = lane & 1;
  const float* fr = feat + ((size_t)b * N_ + n0 + row) * C_ + half * 64;
  v4f reg[16];
  float s = 0.f, ss = 0.f;
#pragma unroll
  for (int i = 0; i < 16; ++i) {
    reg[i] = __builtin_nontemporal_load(((const v4f*)fr) + i);   // TH=NT stream
#pragma unroll
    for (int e = 0; e < 4; ++e) { s += reg[i][e]; ss += reg[i][e]*reg[i][e]; }
  }
  s  += __shfl_xor(s, 1, 32);
  ss += __shfl_xor(ss, 1, 32);
  const float mean = s * (1.f / C_);
  const float var  = ss * (1.f / C_) - mean * mean;
  const float rs   = rsqrtf(var + LN_EPS_);
  __syncthreads();   // wsh/bsh visible
#pragma unroll
  for (int i = 0; i < 16; ++i) {
    int c = half * 64 + i * 4;
#pragma unroll
    for (int e = 0; e < 4; ++e)
      fe[row][c+e] = (_Float16)((reg[i][e] - mean) * rs * wsh[c+e] + bsh[c+e]);
  }
  __syncthreads();

  // --- GEMM: (16 x 128) x (128 x 64) for both Wk and Wv ---
  const int m  = lane & 15;
  const int k0 = (lane & 16) ? 8 : 0;   // start of first 8-elem K-run
  v8f kacc[4], vacc[4];
#pragma unroll
  for (int dc = 0; dc < 4; ++dc) { kacc[dc] = zero8(); vacc[dc] = zero8(); }

#pragma unroll
  for (int kc = 0; kc < 4; ++kc) {
    V16 a;
    a.h[0] = *(const v8h*)&fe[m][kc*32 + k0];
    a.h[1] = *(const v8h*)&fe[m][kc*32 + k0 + 16];
#pragma unroll
    for (int dc = 0; dc < 4; ++dc) {
      const int d = dc*16 + m;
      V16 bk, bv;
      const _Float16* wkr = Wkh + d * C_ + kc*32;
      const _Float16* wvr = Wvh + d * C_ + kc*32;
      bk.h[0] = *(const v8h*)(wkr + k0);
      bk.h[1] = *(const v8h*)(wkr + k0 + 16);
      bv.h[0] = *(const v8h*)(wvr + k0);
      bv.h[1] = *(const v8h*)(wvr + k0 + 16);
      kacc[dc] = __builtin_amdgcn_wmma_f32_16x16x32_f16(
          false, a.v, false, bk.v, (short)0, kacc[dc], false, false);
      vacc[dc] = __builtin_amdgcn_wmma_f32_16x16x32_f16(
          false, a.v, false, bv.v, (short)0, vacc[dc], false, false);
    }
  }

  // --- stage D tiles to LDS (D layout: lane holds col d, regs are rows) ---
  const int mrow0 = (lane & 16) ? 8 : 0;
#pragma unroll
  for (int dc = 0; dc < 4; ++dc) {
    const int d = dc*16 + m;
#pragma unroll
    for (int r = 0; r < 8; ++r) {
      kt[mrow0 + r][d] = (_Float16)kacc[dc][r];
      vt[d][mrow0 + r] = (_Float16)vacc[dc][r];
    }
  }
  __syncthreads();

  // --- coalesced stores (default RT hints: keep k/v resident in L2) ---
  _Float16* kout = kbuf + ((size_t)b * N_ + n0) * D_;
#pragma unroll
  for (int s2 = 0; s2 < 4; ++s2) {
    int idx = lane + s2*32;            // 0..127 -> 16 rows x 8 chunks of 8 halves
    int r = idx >> 3, c = (idx & 7) * 8;
    *(v8h*)(kout + r*D_ + c) = *(const v8h*)&kt[r][c];
  }
#pragma unroll
  for (int s2 = 0; s2 < 4; ++s2) {
    int idx = lane + s2*32;            // 0..127 -> 64 rows x 2 chunks of 8 halves
    int d = idx >> 1, c = (idx & 1) * 8;
    *(v8h*)(vtbuf + ((size_t)b*D_ + d) * N_ + n0 + c) = *(const v8h*)&vt[d][c];
  }
}

// ---------------------------------------------------------------------------
// q = LN(slot) @ Wq^T, padded to 16 rows (rows 8..15 zero), stored f16.
// ---------------------------------------------------------------------------
__global__ __launch_bounds__(256) void q_kernel(
    const float* __restrict__ slot, const float* __restrict__ qlnw,
    const float* __restrict__ qlnb, const float* __restrict__ Wq,
    _Float16* __restrict__ qbuf)
{
  const int t = threadIdx.x;         // 256 = B*16 rows
  const int b = t >> 4, r = t & 15;
  _Float16* out = qbuf + (b*16 + r) * D_;
  if (r >= K_) {
    for (int e = 0; e < D_; ++e) out[e] = (_Float16)0.f;
    return;
  }
  const float* x = slot + (b*K_ + r) * D_;
  float s = 0.f, ss = 0.f;
  for (int d = 0; d < D_; ++d) { s += x[d]; ss += x[d]*x[d]; }
  const float mean = s / D_;
  const float var  = ss / D_ - mean*mean;
  const float rs   = rsqrtf(var + LN_EPS_);
  float xn[D_];
  for (int d = 0; d < D_; ++d) xn[d] = (x[d] - mean) * rs * qlnw[d] + qlnb[d];
  for (int e = 0; e < D_; ++e) {
    float acc = 0.f;
    const float* wr = Wq + e * D_;
    for (int d = 0; d < D_; ++d) acc += wr[d] * xn[d];
    out[e] = (_Float16)acc;
  }
}

// ---------------------------------------------------------------------------
// Fused attention pass: dots = q k^T (WMMA), per-lane softmax over the 8
// slots, accumulate U = sum_n attn*v (WMMA) and S = sum_n attn. One wave
// covers NPW_ columns of N; f32 atomics reduce across waves (target buffer
// is 33 KB -> L2-resident). k/v reads hit L2 thanks to NT feat streaming.
// ---------------------------------------------------------------------------
__global__ __launch_bounds__(32) void attn_kernel(
    const _Float16* __restrict__ qbuf, const _Float16* __restrict__ kbuf,
    const _Float16* __restrict__ vtbuf, float* __restrict__ U,
    float* __restrict__ S, float* __restrict__ attn_out, int write_attn)
{
  __shared__ __align__(16) _Float16 attl[16][32];   // attn staged as A-operand
  const int lane = threadIdx.x;
  const int wpb = N_ / NPW_;
  const int b      = blockIdx.x / wpb;
  const int nstart = (blockIdx.x % wpb) * NPW_;
  const int m  = lane & 15;
  const int k0 = (lane & 16) ? 8 : 0;

  // q A-operand (two K-chunks of 32 over D=64), loaded once
  V16 qa[2];
#pragma unroll
  for (int kc = 0; kc < 2; ++kc) {
    const _Float16* qr = qbuf + (b*16 + m) * D_ + kc*32;
    qa[kc].h[0] = *(const v8h*)(qr + k0);
    qa[kc].h[1] = *(const v8h*)(qr + k0 + 16);
  }

  v8f Uacc[4];
#pragma unroll
  for (int dc = 0; dc < 4; ++dc) Uacc[dc] = zero8();
  float sl[8] = {0.f,0.f,0.f,0.f,0.f,0.f,0.f,0.f};

  for (int i = lane; i < 16*32; i += 32) (&attl[0][0])[i] = (_Float16)0.f;
  __syncthreads();

  for (int ch = 0; ch < CHUNKS_; ++ch) {
    const int nb = nstart + ch*32;
#pragma unroll
    for (int t = 0; t < 2; ++t) {
      v8f dots = zero8();
      const _Float16* krow = kbuf + ((size_t)b * N_ + nb + t*16 + m) * D_;
#pragma unroll
      for (int kc = 0; kc < 2; ++kc) {
        V16 kb;
        kb.h[0] = *(const v8h*)(krow + kc*32 + k0);
        kb.h[1] = *(const v8h*)(krow + kc*32 + k0 + 16);
        dots = __builtin_amdgcn_wmma_f32_16x16x32_f16(
            false, qa[kc].v, false, kb.v, (short)0, dots, false, false);
      }
      // lanes 0..15 hold all 8 slot rows for column n = nb + t*16 + lane:
      // softmax over the slot axis is a pure per-lane 8-element reduction.
      if (lane < 16) {
        float mx = dots[0] * SCALE_;
#pragma unroll
        for (int r = 1; r < 8; ++r) mx = fmaxf(mx, dots[r] * SCALE_);
        float e[8], sum = 0.f;
#pragma unroll
        for (int r = 0; r < 8; ++r) { e[r] = __expf(dots[r]*SCALE_ - mx); sum += e[r]; }
        const float inv = 1.f / sum;
#pragma unroll
        for (int r = 0; r < 8; ++r) {
          const float a = e[r] * inv + EPS_;
          sl[r] += a;
          attl[r][t*16 + lane] = (_Float16)a;
          if (write_attn)   // write-once 8 MB output: NT store, skip L2 residency
            __builtin_nontemporal_store(
                a, &attn_out[((size_t)(b*K_ + r)) * N_ + nb + t*16 + lane]);
        }
      }
    }
    __syncthreads();
    // U += attn (8x32, padded to 16) x v (32 x 64)
    V16 aa;
    aa.h[0] = *(const v8h*)&attl[m][k0];
    aa.h[1] = *(const v8h*)&attl[m][k0 + 16];
#pragma unroll
    for (int dc = 0; dc < 4; ++dc) {
      const _Float16* vr = vtbuf + ((size_t)b*D_ + dc*16 + m) * N_ + nb;
      V16 vb;
      vb.h[0] = *(const v8h*)(vr + k0);
      vb.h[1] = *(const v8h*)(vr + k0 + 16);
      Uacc[dc] = __builtin_amdgcn_wmma_f32_16x16x32_f16(
          false, aa.v, false, vb.v, (short)0, Uacc[dc], false, false);
    }
    __syncthreads();
  }

  if (lane < 16) {
#pragma unroll
    for (int r = 0; r < 8; ++r) atomicAdd(&S[b*K_ + r], sl[r]);
#pragma unroll
    for (int dc = 0; dc < 4; ++dc)
#pragma unroll
      for (int r = 0; r < 8; ++r)
        atomicAdd(&U[(b*K_ + r) * D_ + dc*16 + lane], Uacc[dc][r]);
  }
}

// ---------------------------------------------------------------------------
// updates = U/S ; GRUCell ; LN ; MLP residual. One 64-thread block per
// (b,k) row; thread = output element d.
// ---------------------------------------------------------------------------
__global__ __launch_bounds__(64) void update_kernel(
    const float* __restrict__ U, const float* __restrict__ S,
    float* __restrict__ slotbuf,
    const float* __restrict__ wih, const float* __restrict__ whh,
    const float* __restrict__ bih, const float* __restrict__ bhh,
    const float* __restrict__ rlnw, const float* __restrict__ rlnb,
    const float* __restrict__ w1, const float* __restrict__ b1,
    const float* __restrict__ w2, const float* __restrict__ b2,
    float* __restrict__ out_slot, int last)
{
  __shared__ float upd[D_], prev[D_], sl2[D_], xn[D_], h1[H_];
  const int tid = threadIdx.x;        // 0..63
  const int row = blockIdx.x;         // b*K + k

  upd[tid]  = U[row*D_ + tid] / S[row];
  prev[tid] = slotbuf[row*D_ + tid];
  __syncthreads();

  float ir = bih[tid],       hr = bhh[tid];
  float iz = bih[D_ + tid],  hz = bhh[D_ + tid];
  float in_ = bih[2*D_ + tid], hn = bhh[2*D_ + tid];
  for (int d = 0; d < D_; ++d) {
    ir  += wih[tid*D_ + d]        * upd[d];
    hr  += whh[tid*D_ + d]        * prev[d];
    iz  += wih[(D_+tid)*D_ + d]   * upd[d];
    hz  += whh[(D_+tid)*D_ + d]   * prev[d];
    in_ += wih[(2*D_+tid)*D_ + d] * upd[d];
    hn  += whh[(2*D_+tid)*D_ + d] * prev[d];
  }
  const float r = 1.f / (1.f + __expf(-(ir + hr)));
  const float z = 1.f / (1.f + __expf(-(iz + hz)));
  const float n = tanhf(in_ + r * hn);
  sl2[tid] = (1.f - z) * n + z * prev[tid];
  __syncthreads();

  float s = 0.f, ss = 0.f;
  for (int d = 0; d < D_; ++d) { s += sl2[d]; ss += sl2[d]*sl2[d]; }
  const float mean = s / D_;
  const float var  = ss / D_ - mean*mean;
  const float rs   = rsqrtf(var + LN_EPS_);
  xn[tid] = (sl2[tid] - mean) * rs * rlnw[tid] + rlnb[tid];
  __syncthreads();

  for (int j = tid; j < H_; j += D_) {
    float acc = b1[j];
    for (int d = 0; d < D_; ++d) acc += w1[j*D_ + d] * xn[d];
    h1[j] = fmaxf(acc, 0.f);
  }
  __syncthreads();

  float acc = sl2[tid] + b2[tid];
  for (int h = 0; h < H_; ++h) acc += w2[tid*H_ + h] * h1[h];
  slotbuf[row*D_ + tid] = acc;
  if (last) out_slot[row*D_ + tid] = acc;
}

// ---------------------------------------------------------------------------
extern "C" void kernel_launch(void* const* d_in, const int* in_sizes, int n_in,
                              void* d_out, int out_size, void* d_ws, size_t ws_size,
                              hipStream_t stream)
{
  const float* feat   = (const float*)d_in[0];
  const float* noise  = (const float*)d_in[1];
  const float* mu     = (const float*)d_in[2];
  const float* logsig = (const float*)d_in[3];
  const float* lnw    = (const float*)d_in[4];
  const float* lnb    = (const float*)d_in[5];
  const float* Wk     = (const float*)d_in[6];
  const float* Wv     = (const float*)d_in[7];
  const float* qlnw   = (const float*)d_in[8];
  const float* qlnb   = (const float*)d_in[9];
  const float* Wq     = (const float*)d_in[10];
  const float* wih    = (const float*)d_in[11];
  const float* whh    = (const float*)d_in[12];
  const float* bih    = (const float*)d_in[13];
  const float* bhh    = (const float*)d_in[14];
  const float* rlnw   = (const float*)d_in[15];
  const float* rlnb   = (const float*)d_in[16];
  const float* w1     = (const float*)d_in[17];
  const float* b1     = (const float*)d_in[18];
  const float* w2     = (const float*)d_in[19];
  const float* b2     = (const float*)d_in[20];

  char* ws = (char*)d_ws;
  size_t off = 0;
  _Float16* kbuf  = (_Float16*)(ws + off); off += (size_t)B_*N_*D_*sizeof(_Float16);
  _Float16* vtbuf = (_Float16*)(ws + off); off += (size_t)B_*N_*D_*sizeof(_Float16);
  _Float16* Wkh   = (_Float16*)(ws + off); off += (size_t)D_*C_*sizeof(_Float16);
  _Float16* Wvh   = (_Float16*)(ws + off); off += (size_t)D_*C_*sizeof(_Float16);
  _Float16* qbuf  = (_Float16*)(ws + off); off += (size_t)B_*16*D_*sizeof(_Float16);
  float* slotbuf  = (float*)(ws + off);    off += (size_t)B_*K_*D_*sizeof(float);
  float* U        = (float*)(ws + off);    off += (size_t)B_*K_*D_*sizeof(float);
  float* S        = (float*)(ws + off);    off += (size_t)B_*K_*sizeof(float);

  float* out_slot = (float*)d_out;                 // (B,K,D) = 8192 floats
  float* out_attn = out_slot + B_*K_*D_;           // (B,K,N)

  init_kernel<<<32, 256, 0, stream>>>(noise, mu, logsig, Wk, Wv, slotbuf, Wkh, Wvh);
  lnproj_kernel<<<B_*(N_/16), 32, 0, stream>>>(feat, lnw, lnb, Wkh, Wvh, kbuf, vtbuf);

  for (int it = 0; it < ITERS_; ++it) {
    const int last = (it == ITERS_ - 1);
    zero_us_kernel<<<32, 256, 0, stream>>>(U, S);
    q_kernel<<<1, 256, 0, stream>>>(slotbuf, qlnw, qlnb, Wq, qbuf);
    attn_kernel<<<B_*(N_/NPW_), 32, 0, stream>>>(qbuf, kbuf, vtbuf, U, S,
                                                 out_attn, last);
    update_kernel<<<B_*K_, 64, 0, stream>>>(U, S, slotbuf, wih, whh, bih, bhh,
                                            rlnw, rlnb, w1, b1, w2, b2,
                                            out_slot, last);
  }
}
